// Point2Encoder_14577119002745
// MI455X (gfx1250) — compile-verified, matrix operand order checked by reading, not run on hardware
//
#include <hip/hip_runtime.h>
#include <hip/hip_bf16.h>

// ---------------- problem constants (from reference) ----------------
#define B_   4
#define N_   16384
#define PROJ_ 64
#define M1_  2048
#define M2_  512
#define KNN_ 32
#define C1_  128
#define C2_  256
#define DH_  512
#define DO_  256
#define K1PAD 96    // 3+64=67  -> pad to 96  (mult of 32)
#define K2PAD 160   // 3+128=131-> pad to 160 (mult of 32)

typedef __attribute__((ext_vector_type(16))) _Float16 hvec16;
typedef __attribute__((ext_vector_type(8)))  _Float16 hvec8;
typedef __attribute__((ext_vector_type(8)))  float    fvec8;

// =====================================================================
// 0. projection: feats0[b][n][c] = xyz[b][n][:] @ Wp + bp   (f16 out)
// =====================================================================
__global__ void proj_kernel(const float* __restrict__ xyz,
                            const float* __restrict__ Wp,
                            const float* __restrict__ bp,
                            _Float16* __restrict__ f0, int total)
{
    int i = blockIdx.x * blockDim.x + threadIdx.x;
    if (i >= total) return;
    int c = i & (PROJ_ - 1);
    size_t p = (size_t)i >> 6;
    const float* x = xyz + p * 3;
    float v = x[0] * Wp[c] + x[1] * Wp[PROJ_ + c] + x[2] * Wp[2 * PROJ_ + c] + bp[c];
    f0[i] = (_Float16)v;
}

// =====================================================================
// 1. weight convert: W [K][N] f32 -> WT [N][Kpad] f16 (zero-padded)
// =====================================================================
__global__ void wconv_kernel(const float* __restrict__ W,
                             _Float16* __restrict__ WT,
                             int K, int N, int Kpad)
{
    int i = blockIdx.x * blockDim.x + threadIdx.x;
    if (i >= N * Kpad) return;
    int n = i / Kpad, k = i % Kpad;
    WT[i] = (k < K) ? (_Float16)W[(size_t)k * N + n] : (_Float16)0.0f;
}

// =====================================================================
// 2. FPS: one workgroup per batch; entire point cloud + dmin in LDS
//    (n*16 + 1024*8 bytes dynamic LDS; <=270KB fits the 320KB WGP LDS)
// =====================================================================
__global__ __launch_bounds__(1024)
void fps_kernel(const float* __restrict__ pts, int n, int M,
                int* __restrict__ idx_out, float* __restrict__ centers)
{
    extern __shared__ char smem[];
    float* sx = (float*)smem;
    float* sy = sx + n;
    float* sz = sy + n;
    float* sd = sz + n;
    float* rv = sd + n;
    int*   ri = (int*)(rv + 1024);
    __shared__ int s_last;

    int b = blockIdx.x, tid = threadIdx.x;
    const float* P = pts + (size_t)b * n * 3;
    for (int j = tid; j < n; j += 1024) {
        sx[j] = P[j * 3 + 0];
        sy[j] = P[j * 3 + 1];
        sz[j] = P[j * 3 + 2];
        sd[j] = 1.0e10f;
    }
    if (tid == 0) s_last = 0;
    __syncthreads();

    for (int it = 0; it < M; ++it) {
        int last = s_last;
        if (tid == 0) {
            idx_out[(size_t)b * M + it] = last;
            float* c = centers + ((size_t)b * M + it) * 3;
            c[0] = sx[last]; c[1] = sy[last]; c[2] = sz[last];
        }
        float lx = sx[last], ly = sy[last], lz = sz[last];
        float bv = -1.0f; int bi = 0;
        for (int j = tid; j < n; j += 1024) {
            float dx = sx[j] - lx, dy = sy[j] - ly, dz = sz[j] - lz;
            float d = dx * dx + dy * dy + dz * dz;
            float m = fminf(sd[j], d);
            sd[j] = m;
            if (m > bv) { bv = m; bi = j; }
        }
        rv[tid] = bv; ri[tid] = bi;
        __syncthreads();
        for (int s = 512; s > 0; s >>= 1) {
            if (tid < s) {
                float ov = rv[tid + s]; int oi = ri[tid + s];
                if (ov > rv[tid] || (ov == rv[tid] && oi < ri[tid])) {
                    rv[tid] = ov; ri[tid] = oi;
                }
            }
            __syncthreads();
        }
        if (tid == 0) s_last = ri[0];
        __syncthreads();
    }
}

// =====================================================================
// 3. kNN top-32: one thread per center, 32-entry max-heap in LDS,
//    points streamed through a 2048-pt LDS tile (+prefetch next tile)
// =====================================================================
#define KTILE 2048
__global__ __launch_bounds__(256)
void knn_kernel(const float* __restrict__ pts,     // [B][n][3]
                const float* __restrict__ centers, // [B][M][3]
                int n, int M, int* __restrict__ nidx) // [B][M][32]
{
    extern __shared__ char smem[];
    float* tx = (float*)smem;
    float* ty = tx + KTILE;
    float* tz = ty + KTILE;
    float* hd = tz + KTILE;                // 256*32 f32
    int*   hi = (int*)(hd + 256 * KNN_);   // 256*32 i32

    int b = blockIdx.y, tid = threadIdx.x;
    int c = blockIdx.x * 256 + tid;
    const float* cc = centers + ((size_t)b * M + c) * 3;
    float cx = cc[0], cy = cc[1], cz = cc[2];

    float* md = hd + tid * KNN_;
    int*   mi = hi + tid * KNN_;
#pragma unroll
    for (int i = 0; i < KNN_; ++i) { md[i] = 3.0e38f; mi[i] = 0; }

    const float* P = pts + (size_t)b * n * 3;
    for (int t0 = 0; t0 < n; t0 += KTILE) {
        int cnt = min(KTILE, n - t0);
        if (t0 + KTILE < n)
            __builtin_prefetch(P + (size_t)(t0 + KTILE) * 3, 0, 1);
        __syncthreads();
        for (int j = tid; j < cnt; j += 256) {
            tx[j] = P[(size_t)(t0 + j) * 3 + 0];
            ty[j] = P[(size_t)(t0 + j) * 3 + 1];
            tz[j] = P[(size_t)(t0 + j) * 3 + 2];
        }
        __syncthreads();
        for (int j = 0; j < cnt; ++j) {
            float dx = tx[j] - cx, dy = ty[j] - cy, dz = tz[j] - cz;
            float d = dx * dx + dy * dy + dz * dz;
            if (d < md[0]) {            // replace current worst, sift down
                int p = 0;
                for (;;) {
                    int l = 2 * p + 1;
                    if (l >= KNN_) break;
                    int g = l; float gv = md[l];
                    if (l + 1 < KNN_ && md[l + 1] > gv) { g = l + 1; gv = md[l + 1]; }
                    if (gv <= d) break;
                    md[p] = gv; mi[p] = mi[g]; p = g;
                }
                md[p] = d; mi[p] = t0 + j;
            }
        }
    }
    int* out = nidx + ((size_t)b * M + c) * KNN_;
#pragma unroll
    for (int i = 0; i < KNN_; ++i) out[i] = mi[i];
}

// =====================================================================
// 4. gather/build grouped-MLP input rows: g[row] = [rel(3) | feats(Cin) | 0pad]
// =====================================================================
__global__ void build_g_kernel(const float* __restrict__ pts,      // [B][n][3]
                               const float* __restrict__ centers,  // [B][M][3]
                               const _Float16* __restrict__ feats, // [B][n][Cin]
                               const int* __restrict__ nidx,       // [B][M][32]
                               int n, int M, int Cin, int Kpad,
                               long long row0, _Float16* __restrict__ g,
                               int rows)
{
    int rl = blockIdx.x * blockDim.x + threadIdx.x;
    if (rl >= rows) return;
    long long r = row0 + rl;
    int b = (int)(r / ((long long)M * KNN_));
    long long rem = r % ((long long)M * KNN_);
    int m = (int)(rem / KNN_);
    int j = nidx[r];

    const float* pp = pts + ((size_t)b * n + j) * 3;
    const float* ct = centers + ((size_t)b * M + m) * 3;
    _Float16* o = g + (size_t)rl * Kpad;
    o[0] = (_Float16)(pp[0] - ct[0]);
    o[1] = (_Float16)(pp[1] - ct[1]);
    o[2] = (_Float16)(pp[2] - ct[2]);
    const _Float16* f = feats + ((size_t)b * n + j) * Cin;
    for (int c = 0; c < Cin; ++c) o[3 + c] = f[c];
    for (int c = 3 + Cin; c < Kpad; ++c) o[c] = (_Float16)0.0f;
}

// =====================================================================
// 5. WMMA GEMM: C[M][N] = act(A[M][K] * BT[N][K]^T + bias)
//    One wave per 16x16 tile; A/B fragments follow the CDNA5 16-bit
//    A-matrix layout: lane = row(m or n), halves 0..7 -> K = 8*(lane/16)+i,
//    halves 8..15 -> K = 16 + 8*(lane/16) + (i-8)  => two 16B runs/lane.
// =====================================================================
__device__ inline hvec16 load_frag16(const _Float16* __restrict__ tile,
                                     int ld, int lane)
{
    int row = lane & 15;
    int off = (lane >> 4) * 8;
    const _Float16* p = tile + (size_t)row * ld + off;
    hvec8 lo = *(const hvec8*)p;
    hvec8 hi = *(const hvec8*)(p + 16);
    hvec16 r;
#pragma unroll
    for (int i = 0; i < 8; ++i) { r[i] = lo[i]; r[i + 8] = hi[i]; }
    return r;
}

template <bool RELU, typename OutT>
__global__ __launch_bounds__(256)
void gemm_wmma_kernel(const _Float16* __restrict__ A,   // [Mrows][K]
                      const _Float16* __restrict__ BT,  // [N][K]
                      const float* __restrict__ bias,   // [N]
                      OutT* __restrict__ C,             // [Mrows][N]
                      int N, int K)
{
    int lane = threadIdx.x & 31;
    int wv   = threadIdx.x >> 5;
    int tm   = blockIdx.x;
    int tn   = blockIdx.y * 8 + wv;

    const _Float16* Abase = A  + (size_t)tm * 16 * K;
    const _Float16* Bbase = BT + (size_t)tn * 16 * K;

    fvec8 acc = {0.f, 0.f, 0.f, 0.f, 0.f, 0.f, 0.f, 0.f};
    for (int k = 0; k < K; k += 32) {
        hvec16 a = load_frag16(Abase + k, K, lane);
        hvec16 b = load_frag16(Bbase + k, K, lane);
        acc = __builtin_amdgcn_wmma_f32_16x16x32_f16(
            false, a, false, b, (short)0, acc, false, false);
    }
    int n    = tn * 16 + (lane & 15);
    int mrow = tm * 16 + (lane >> 4) * 8;
    float bv = bias[n];
#pragma unroll
    for (int v = 0; v < 8; ++v) {
        float x = acc[v] + bv;
        if (RELU) x = fmaxf(x, 0.0f);
        C[(size_t)(mrow + v) * N + n] = (OutT)x;
    }
}

// =====================================================================
// 6. max-pool over groups of 32 consecutive rows (the K neighbors)
// =====================================================================
template <typename OutT>
__global__ void maxpool32_kernel(const _Float16* __restrict__ h, int C,
                                 long long grp0, OutT* __restrict__ out)
{
    long long g = grp0 + blockIdx.x;
    int c = threadIdx.x;
    const _Float16* p = h + (size_t)blockIdx.x * KNN_ * C + c;
    float best = -3.0e38f;
#pragma unroll
    for (int k = 0; k < KNN_; ++k) best = fmaxf(best, (float)p[(size_t)k * C]);
    out[(size_t)g * C + c] = (OutT)best;
}

// =====================================================================
// 7. global max||mean pooling and dense head
// =====================================================================
__global__ void global_pool_kernel(const float* __restrict__ feats2, // [B][512][256]
                                   float* __restrict__ gg)           // [B][512]
{
    int b = blockIdx.x, c = threadIdx.x;
    const float* p = feats2 + (size_t)b * M2_ * C2_ + c;
    float mx = -3.0e38f, sm = 0.f;
    for (int m = 0; m < M2_; ++m) {
        float v = p[(size_t)m * C2_];
        mx = fmaxf(mx, v); sm += v;
    }
    gg[b * 2 * C2_ + c]       = mx;
    gg[b * 2 * C2_ + C2_ + c] = sm * (1.0f / (float)M2_);
}

__global__ void head1_kernel(const float* __restrict__ gg,
                             const float* __restrict__ Wd1,
                             const float* __restrict__ bd1,
                             float* __restrict__ h)
{
    int b = blockIdx.x, t = threadIdx.x;
    const float* x = gg + b * DH_;
    float acc = bd1[t];
    for (int k = 0; k < 2 * C2_; ++k) acc += x[k] * Wd1[(size_t)k * DH_ + t];
    h[b * DH_ + t] = fmaxf(acc, 0.0f);
}

__global__ void head2_kernel(const float* __restrict__ h,
                             const float* __restrict__ Wd2,
                             const float* __restrict__ bd2,
                             float* __restrict__ out)
{
    int b = blockIdx.x, t = threadIdx.x;
    const float* x = h + b * DH_;
    float acc = bd2[t];
    for (int k = 0; k < DH_; ++k) acc += x[k] * Wd2[(size_t)k * DO_ + t];
    out[b * DO_ + t] = acc;
}

// =====================================================================
// launch
// =====================================================================
extern "C" void kernel_launch(void* const* d_in, const int* in_sizes, int n_in,
                              void* d_out, int out_size, void* d_ws, size_t ws_size,
                              hipStream_t stream)
{
    (void)in_sizes; (void)n_in; (void)out_size; (void)ws_size;

    const float* xyz = (const float*)d_in[0];
    const float* Wp  = (const float*)d_in[1];
    const float* bp  = (const float*)d_in[2];
    const float* W1a = (const float*)d_in[3];
    const float* b1a = (const float*)d_in[4];
    const float* W1b = (const float*)d_in[5];
    const float* b1b = (const float*)d_in[6];
    const float* W2a = (const float*)d_in[7];
    const float* b2a = (const float*)d_in[8];
    const float* W2b = (const float*)d_in[9];
    const float* b2b = (const float*)d_in[10];
    const float* Wd1 = (const float*)d_in[11];
    const float* bd1 = (const float*)d_in[12];
    const float* Wd2 = (const float*)d_in[13];
    const float* bd2 = (const float*)d_in[14];
    float* outp = (float*)d_out;

    // ---- bump allocator over d_ws ----
    char* base = (char*)d_ws;
    size_t off = 0;
    auto alloc = [&](size_t bytes) -> void* {
        off = (off + 255) & ~(size_t)255;
        void* p = base + off;
        off += bytes;
        return p;
    };

    _Float16* f0      = (_Float16*)alloc((size_t)B_ * N_ * PROJ_ * 2);
    _Float16* W1aT    = (_Float16*)alloc((size_t)C1_ * K1PAD * 2);
    _Float16* W1bT    = (_Float16*)alloc((size_t)C1_ * C1_ * 2);
    _Float16* W2aT    = (_Float16*)alloc((size_t)C2_ * K2PAD * 2);
    _Float16* W2bT    = (_Float16*)alloc((size_t)C2_ * C2_ * 2);
    int*      idx1    = (int*)alloc((size_t)B_ * M1_ * 4);
    float*    cent1   = (float*)alloc((size_t)B_ * M1_ * 3 * 4);
    int*      knn1    = (int*)alloc((size_t)B_ * M1_ * KNN_ * 4);
    int*      idx2    = (int*)alloc((size_t)B_ * M2_ * 4);
    float*    cent2   = (float*)alloc((size_t)B_ * M2_ * 3 * 4);
    int*      knn2    = (int*)alloc((size_t)B_ * M2_ * KNN_ * 4);
    _Float16* feats1  = (_Float16*)alloc((size_t)B_ * M1_ * C1_ * 2);
    float*    feats2  = (float*)alloc((size_t)B_ * M2_ * C2_ * 4);
    const int CH = 32768; // rows per chunk (mult of 32 and 16)
    _Float16* gbuf    = (_Float16*)alloc((size_t)CH * K2PAD * 2);
    _Float16* hA      = (_Float16*)alloc((size_t)CH * C2_ * 2);
    _Float16* hB      = (_Float16*)alloc((size_t)CH * C2_ * 2);
    float*    gg      = (float*)alloc((size_t)B_ * 2 * C2_ * 4);
    float*    hh      = (float*)alloc((size_t)B_ * DH_ * 4);

    // raise dynamic-LDS limits (no-op if unsupported)
    (void)hipFuncSetAttribute(reinterpret_cast<const void*>(fps_kernel),
                              hipFuncAttributeMaxDynamicSharedMemorySize,
                              N_ * 16 + 1024 * 8);
    (void)hipFuncSetAttribute(reinterpret_cast<const void*>(knn_kernel),
                              hipFuncAttributeMaxDynamicSharedMemorySize,
                              KTILE * 12 + 256 * KNN_ * 8);

    // ---- stage 0: projection + weight conversion ----
    {
        int total = B_ * N_ * PROJ_;
        proj_kernel<<<(total + 255) / 256, 256, 0, stream>>>(xyz, Wp, bp, f0, total);
        wconv_kernel<<<(C1_ * K1PAD + 255) / 256, 256, 0, stream>>>(W1a, W1aT, 67, C1_, K1PAD);
        wconv_kernel<<<(C1_ * C1_ + 255) / 256, 256, 0, stream>>>(W1b, W1bT, C1_, C1_, C1_);
        wconv_kernel<<<(C2_ * K2PAD + 255) / 256, 256, 0, stream>>>(W2a, W2aT, 131, C2_, K2PAD);
        wconv_kernel<<<(C2_ * C2_ + 255) / 256, 256, 0, stream>>>(W2b, W2bT, C2_, C2_, C2_);
    }

    // ---- stage 1: FPS + kNN level 1 ----
    fps_kernel<<<B_, 1024, N_ * 16 + 1024 * 8, stream>>>(xyz, N_, M1_, idx1, cent1);
    knn_kernel<<<dim3(M1_ / 256, B_), 256, KTILE * 12 + 256 * KNN_ * 8, stream>>>(
        xyz, cent1, N_, M1_, knn1);

    // ---- stage 2: grouped MLP level 1 (chunked WMMA GEMMs) ----
    {
        long long total = (long long)B_ * M1_ * KNN_; // 262144
        for (long long r0 = 0; r0 < total; r0 += CH) {
            build_g_kernel<<<CH / 256, 256, 0, stream>>>(
                xyz, cent1, f0, knn1, N_, M1_, PROJ_, K1PAD, r0, gbuf, CH);
            gemm_wmma_kernel<true, _Float16><<<dim3(CH / 16, 1), 256, 0, stream>>>(
                gbuf, W1aT, b1a, hA, C1_, K1PAD);
            gemm_wmma_kernel<true, _Float16><<<dim3(CH / 16, 1), 256, 0, stream>>>(
                hA, W1bT, b1b, hB, C1_, C1_);
            maxpool32_kernel<_Float16><<<CH / 32, C1_, 0, stream>>>(
                hB, C1_, r0 / 32, feats1);
        }
    }

    // ---- stage 3: FPS + kNN level 2 (on level-1 centers) ----
    fps_kernel<<<B_, 1024, M1_ * 16 + 1024 * 8, stream>>>(cent1, M1_, M2_, idx2, cent2);
    knn_kernel<<<dim3(M2_ / 256, B_), 256, KTILE * 12 + 256 * KNN_ * 8, stream>>>(
        cent1, cent2, M1_, M2_, knn2);

    // ---- stage 4: grouped MLP level 2 ----
    {
        long long total = (long long)B_ * M2_ * KNN_; // 65536
        for (long long r0 = 0; r0 < total; r0 += CH) {
            build_g_kernel<<<CH / 256, 256, 0, stream>>>(
                cent1, cent2, feats1, knn2, M1_, M2_, C1_, K2PAD, r0, gbuf, CH);
            gemm_wmma_kernel<true, _Float16><<<dim3(CH / 16, 2), 256, 0, stream>>>(
                gbuf, W2aT, b2a, hA, C2_, K2PAD);
            gemm_wmma_kernel<true, _Float16><<<dim3(CH / 16, 2), 256, 0, stream>>>(
                hA, W2bT, b2b, hB, C2_, C2_);
            maxpool32_kernel<float><<<CH / 32, C2_, 0, stream>>>(
                hB, C2_, r0 / 32, feats2);
        }
    }

    // ---- stage 5: global pooling + dense head ----
    global_pool_kernel<<<B_, C2_, 0, stream>>>(feats2, gg);
    head1_kernel<<<B_, DH_, 0, stream>>>(gg, Wd1, bd1, hh);
    head2_kernel<<<B_, DO_, 0, stream>>>(hh, Wd2, bd2, outp);
}
